// BoundaryModule_38422777430159
// MI455X (gfx1250) — matrix operand order repaired and checked by reading, not compiled, not run on hardware
//
#include <hip/hip_runtime.h>
#include <hip/hip_bf16.h>
#include <math.h>

// ---------------- problem constants ----------------
#define TT   256
#define NN   10
#define DD   40
#define BB   4
#define CIN  256
#define DIM0 512
#define DIM1 128
#define NPIX (BB*DD*TT)        // 40960 output pixels
#define PADD (DD+2)            // 42
#define PADT (TT+2)            // 258

// ---------------- bf16 / WMMA types ----------------
typedef __bf16 bf16;
typedef bf16  bf16x8 __attribute__((ext_vector_type(8)));
typedef bf16  v16bf  __attribute__((ext_vector_type(16)));
typedef float v8f    __attribute__((ext_vector_type(8)));

__device__ __forceinline__ v16bf concat8(bf16x8 lo, bf16x8 hi) {
  v16bf r;
#pragma unroll
  for (int i = 0; i < 8; ++i) { r[i] = lo[i]; r[i + 8] = hi[i]; }
  return r;
}

// A fragment (16x32 bf16, MxK): lane holds row (lane&15); K chunks at
// kb + 8*(lane>>4) and kb + 8*(lane>>4) + 16  (ISA 7.12.2, 16-bit A 16x32)
__device__ __forceinline__ v16bf load_a(const bf16* row, int kb, int half) {
  bf16x8 lo = *(const bf16x8*)(row + kb + 8 * half);
  bf16x8 hi = *(const bf16x8*)(row + kb + 8 * half + 16);
  return concat8(lo, hi);
}
// B fragment (32x16 bf16, KxN) from N-major storage: lane holds column
// (lane&15), K chunks at kb + 8*(lane>>4) and kb + 8*(lane>>4) + 16
__device__ __forceinline__ v16bf load_b(const bf16* col, int kb, int half) {
  bf16x8 lo = *(const bf16x8*)(col + kb + 8 * half);
  bf16x8 hi = *(const bf16x8*)(col + kb + 8 * half + 16);
  return concat8(lo, hi);
}

// ---------------- workspace layout (all 256B-aligned naturally) -------------
#define W0T_OFF   ((size_t)0)                                 // (10,512,256) bf16
#define FEAT_OFF  (W0T_OFF  + (size_t)NN*DIM0*CIN*2)          // (1024,256) bf16
#define W1B_OFF   (FEAT_OFF + (size_t)BB*TT*CIN*2)            // (128,512) bf16
#define W2T_OFF   (W1B_OFF  + (size_t)DIM1*DIM0*2)            // (9,128,128) bf16
#define SIL_OFF   (W2T_OFF  + (size_t)9*DIM1*DIM1*2)          // (10,40,256) i32
#define SIR_OFF   (SIL_OFF  + (size_t)NN*DD*TT*4)
#define SWL_OFF   (SIR_OFF  + (size_t)NN*DD*TT*4)             // f32
#define SWR_OFF   (SWL_OFF  + (size_t)NN*DD*TT*4)
#define Y_OFF     (SWR_OFF  + (size_t)NN*DD*TT*4)             // (4,10,256,512) f32
#define X_OFF     (Y_OFF    + (size_t)BB*NN*TT*DIM0*4)        // (40960,512) bf16
#define ZPAD_OFF  (X_OFF    + (size_t)NPIX*DIM0*2)            // (4,42,258,128) bf16
#define COUT_OFF  (ZPAD_OFF + (size_t)BB*PADD*PADT*DIM1*2)    // (40960,128) bf16

// ---------------- kernels ----------------
__global__ void zero_u32_kernel(unsigned* p, int n) {
  for (int i = blockIdx.x * blockDim.x + threadIdx.x; i < n;
       i += gridDim.x * blockDim.x) p[i] = 0u;
}

// Recompute the bilinear sampling taps (faithful float64 replication of
// build_smp_weight) instead of reading the 100MB dense smp_weight.
__global__ void samp_kernel(int* iL, int* iR, float* wl, float* wr) {
  int s = blockIdx.x * blockDim.x + threadIdx.x;
  if (s >= NN * DD * TT) return;
  int t = s & (TT - 1);
  int d = (s >> 8) % DD;
  int n = s / (DD * TT);
  int L = 0, R = 0; float a = 0.f, c = 0.f;
  if (t < TT - 1 && d < TT - 1 - t) {            // i < T-1, j < T-1-i
    double xmin = (double)t;
    double xmax = (double)(d + 1);
    double length   = xmax - xmin;
    double xmin_ext = xmin - length * 0.5;
    double xmax_ext = xmax + length * 0.5;
    double bin = (xmax_ext - xmin_ext) / (double)(NN - 1);
    double xp  = xmin_ext + (double)n * bin;
    if (!(xp < 0.0 || xp > (double)(TT - 1))) {
      double fl = floor(xp), ce = ceil(xp);
      L = (int)fl; R = (int)ce;
      a = (float)(1.0 - (xp - fl));
      c = (float)(1.0 - (ce - xp));
    }
  }
  iL[s] = L; iR[s] = R; wl[s] = a; wr[s] = c;
}

// layout converts (fp32 -> bf16, GEMM-friendly layouts)
__global__ void cvt_w0_kernel(const float* w0, bf16* w0t) {      // (o,c,n)->(n,o,c)
  int i = blockIdx.x * blockDim.x + threadIdx.x;
  if (i >= NN * DIM0 * CIN) return;
  int c = i & (CIN - 1); int o = (i >> 8) % DIM0; int n = i / (DIM0 * CIN);
  w0t[i] = (bf16)w0[((size_t)o * CIN + c) * NN + n];
}
__global__ void cvt_feat_kernel(const float* f, bf16* ft) {      // (b,c,t)->(b,t,c)
  int i = blockIdx.x * blockDim.x + threadIdx.x;
  if (i >= BB * TT * CIN) return;
  int c = i & (CIN - 1); int row = i >> 8;       // row = b*T + tau
  int b = row >> 8; int tau = row & (TT - 1);
  ft[i] = (bf16)f[((size_t)b * CIN + c) * TT + tau];
}
__global__ void cvt_w1_kernel(const float* w1, bf16* w1b) {      // same layout
  int i = blockIdx.x * blockDim.x + threadIdx.x;
  if (i >= DIM1 * DIM0) return;
  w1b[i] = (bf16)w1[i];
}
__global__ void cvt_w2_kernel(const float* w2, bf16* w2t) {      // (p,q,kh,kw)->(kh,kw,p,q)
  int i = blockIdx.x * blockDim.x + threadIdx.x;
  if (i >= 9 * DIM1 * DIM1) return;
  int q = i & (DIM1 - 1); int p = (i >> 7) % DIM1; int f = i / (DIM1 * DIM1);
  int kh = f / 3, kw = f % 3;
  w2t[i] = (bf16)w2[(((size_t)p * DIM1 + q) * 3 + kh) * 3 + kw];
}

// Stage 1: y[b,n,tau,o] = sum_c w0[o,c,n]*feat[b,c,tau]   (10x GEMM 512x1024x256)
__global__ void gemm_y_kernel(const bf16* __restrict__ w0t,
                              const bf16* __restrict__ ft,
                              float* __restrict__ y) {
  int wid  = blockIdx.x * (blockDim.x >> 5) + (threadIdx.x >> 5);
  int lane = threadIdx.x & 31, half = lane >> 4, l15 = lane & 15;
  int nt = wid & 63;              // N tiles: 1024/16
  int mt = (wid >> 6) & 31;       // M tiles: 512/16
  int nn = wid >> 11;             // n slice: 10
  const bf16* arow = w0t + ((size_t)(nn * DIM0 + mt * 16 + l15)) * CIN;
  const bf16* bcol = ft  + ((size_t)(nt * 16 + l15)) * CIN;
  v8f acc = {};
#pragma unroll
  for (int kt = 0; kt < CIN / 32; ++kt)
    acc = __builtin_amdgcn_wmma_f32_16x16x32_bf16(
        false, load_a(arow, kt * 32, half),
        false, load_b(bcol, kt * 32, half),
        (short)0, acc, false, false);
  int col = nt * 16 + l15;        // (b,tau)
  int b = col >> 8, tau = col & (TT - 1);
  float* yb = y + ((size_t)(b * NN + nn) * TT + tau) * DIM0;
#pragma unroll
  for (int r = 0; r < 8; ++r) yb[mt * 16 + r + 8 * half] = acc[r];
}

// Stage 2: x[pixel,o] = leaky( sum_n wl*y[..L] + wr*y[..R] + b0 )  (gather-lerp)
__global__ void sample_x_kernel(const float* __restrict__ y,
                                const int* __restrict__ iL, const int* __restrict__ iR,
                                const float* __restrict__ wl, const float* __restrict__ wr,
                                const float* __restrict__ b0, bf16* __restrict__ x) {
  int pixel = blockIdx.x;                 // (b,d,t)
  int b = pixel / (DD * TT); int rem = pixel % (DD * TT);
  int d = rem >> 8; int t = rem & (TT - 1);
  int o0 = threadIdx.x, o1 = threadIdx.x + 256;
  float a0 = 0.f, a1 = 0.f;
#pragma unroll
  for (int n = 0; n < NN; ++n) {
    int s = (n * DD + d) * TT + t;        // block-uniform -> scalar loads
    int L = iL[s], R = iR[s];
    float wa = wl[s], wb = wr[s];
    const float* yl = y + ((size_t)(b * NN + n) * TT + L) * DIM0;
    const float* yr = y + ((size_t)(b * NN + n) * TT + R) * DIM0;
    a0 += wa * yl[o0] + wb * yr[o0];
    a1 += wa * yl[o1] + wb * yr[o1];
  }
  a0 += b0[o0]; a1 += b0[o1];
  a0 = a0 > 0.f ? a0 : 0.01f * a0;
  a1 = a1 > 0.f ? a1 : 0.01f * a1;
  bf16* xp = x + (size_t)pixel * DIM0;
  xp[o0] = (bf16)a0; xp[o1] = (bf16)a1;
}

// Stage 3: 1x1 conv: z[p,pixel] = leaky(w1 @ x + b1), stored zero-padded NHWC
__global__ void gemm_z_kernel(const bf16* __restrict__ w1b,
                              const bf16* __restrict__ x,
                              const float* __restrict__ b1,
                              bf16* __restrict__ zpad) {
  int wid  = blockIdx.x * (blockDim.x >> 5) + (threadIdx.x >> 5);
  int lane = threadIdx.x & 31, half = lane >> 4, l15 = lane & 15;
  int nt = wid % (NPIX / 16);     // 2560 N tiles
  int mt = wid / (NPIX / 16);     // 8 M tiles
  const bf16* arow = w1b + (size_t)(mt * 16 + l15) * DIM0;
  const bf16* bcol = x   + (size_t)(nt * 16 + l15) * DIM0;
  v8f acc = {};
#pragma unroll
  for (int kt = 0; kt < DIM0 / 32; ++kt)
    acc = __builtin_amdgcn_wmma_f32_16x16x32_bf16(
        false, load_a(arow, kt * 32, half),
        false, load_b(bcol, kt * 32, half),
        (short)0, acc, false, false);
  int pixel = nt * 16 + l15;
  int b = pixel / (DD * TT); int rem = pixel % (DD * TT);
  int d = rem >> 8; int t = rem & (TT - 1);
  bf16* zb = zpad + ((size_t)(b * PADD + d + 1) * PADT + (t + 1)) * DIM1;
#pragma unroll
  for (int r = 0; r < 8; ++r) {
    int p = mt * 16 + r + 8 * half;
    float v = acc[r] + b1[p];
    zb[p] = (bf16)(v > 0.f ? v : 0.01f * v);
  }
}

// Stage 4: 3x3 SAME conv as implicit GEMM: 9 shifted K=128 WMMA accumulations
__global__ void conv3x3_kernel(const bf16* __restrict__ w2t,
                               const bf16* __restrict__ zpad,
                               const float* __restrict__ b2,
                               bf16* __restrict__ cout) {
  int wid  = blockIdx.x * (blockDim.x >> 5) + (threadIdx.x >> 5);
  int lane = threadIdx.x & 31, half = lane >> 4, l15 = lane & 15;
  int nt = wid % (NPIX / 16);
  int mt = wid / (NPIX / 16);
  int pixel = nt * 16 + l15;
  int b = pixel / (DD * TT); int rem = pixel % (DD * TT);
  int d = rem >> 8; int t = rem & (TT - 1);
  v8f acc = {};
#pragma unroll
  for (int kh = 0; kh < 3; ++kh)
#pragma unroll
    for (int kw = 0; kw < 3; ++kw) {
      const bf16* arow = w2t + ((size_t)((kh * 3 + kw) * DIM1) + mt * 16 + l15) * DIM1;
      const bf16* bcol = zpad + ((size_t)(b * PADD + d + kh) * PADT + (t + kw)) * DIM1;
#pragma unroll
      for (int kt = 0; kt < DIM1 / 32; ++kt)
        acc = __builtin_amdgcn_wmma_f32_16x16x32_bf16(
            false, load_a(arow, kt * 32, half),
            false, load_b(bcol, kt * 32, half),
            (short)0, acc, false, false);
    }
  bf16* ob = cout + (size_t)pixel * DIM1;
#pragma unroll
  for (int r = 0; r < 8; ++r) {
    int p = mt * 16 + r + 8 * half;
    float v = acc[r] + b2[p];
    ob[p] = (bf16)(v > 0.f ? v : 0.01f * v);
  }
}

// Stage 5: 1x(128) head + sigmoid -> (B,D,T) fp32
__global__ void head_kernel(const bf16* __restrict__ cout,
                            const float* __restrict__ w3,
                            const float* __restrict__ b3,
                            float* __restrict__ out) {
  int pixel = blockIdx.x * blockDim.x + threadIdx.x;
  if (pixel >= NPIX) return;
  const bf16* c = cout + (size_t)pixel * DIM1;
  float acc = 0.f;
#pragma unroll
  for (int q = 0; q < DIM1; ++q) acc += (float)c[q] * w3[q];
  acc += b3[0];
  out[pixel] = 1.f / (1.f + expf(-acc));
}

// ---------------- launcher ----------------
extern "C" void kernel_launch(void* const* d_in, const int* in_sizes, int n_in,
                              void* d_out, int out_size, void* d_ws, size_t ws_size,
                              hipStream_t stream) {
  const float* feature = (const float*)d_in[0];
  /* d_in[1] = smp_weight: unused — taps recomputed on device */
  const float* w0 = (const float*)d_in[2];
  const float* b0 = (const float*)d_in[3];
  const float* w1 = (const float*)d_in[4];
  const float* b1 = (const float*)d_in[5];
  const float* w2 = (const float*)d_in[6];
  const float* b2 = (const float*)d_in[7];
  const float* w3 = (const float*)d_in[8];
  const float* b3 = (const float*)d_in[9];
  float* out = (float*)d_out;
  char* ws = (char*)d_ws;

  bf16*  w0t  = (bf16*)(ws + W0T_OFF);
  bf16*  ft   = (bf16*)(ws + FEAT_OFF);
  bf16*  w1b  = (bf16*)(ws + W1B_OFF);
  bf16*  w2t  = (bf16*)(ws + W2T_OFF);
  int*   iL   = (int*)(ws + SIL_OFF);
  int*   iR   = (int*)(ws + SIR_OFF);
  float* swl  = (float*)(ws + SWL_OFF);
  float* swr  = (float*)(ws + SWR_OFF);
  float* y    = (float*)(ws + Y_OFF);
  bf16*  x    = (bf16*)(ws + X_OFF);
  bf16*  zpad = (bf16*)(ws + ZPAD_OFF);
  bf16*  cout = (bf16*)(ws + COUT_OFF);

  // zero the padded conv input (border must be 0)
  int zn = (int)((size_t)BB * PADD * PADT * DIM1 * 2 / 4);
  zero_u32_kernel<<<1024, 256, 0, stream>>>((unsigned*)zpad, zn);
  // sampling taps (replaces the 100MB smp_weight GEMM)
  samp_kernel<<<(NN * DD * TT + 255) / 256, 256, 0, stream>>>(iL, iR, swl, swr);
  // bf16 layout converts
  cvt_w0_kernel  <<<(NN * DIM0 * CIN + 255) / 256, 256, 0, stream>>>(w0, w0t);
  cvt_feat_kernel<<<(BB * TT * CIN   + 255) / 256, 256, 0, stream>>>(feature, ft);
  cvt_w1_kernel  <<<(DIM1 * DIM0     + 255) / 256, 256, 0, stream>>>(w1, w1b);
  cvt_w2_kernel  <<<(9 * DIM1 * DIM1 + 255) / 256, 256, 0, stream>>>(w2, w2t);
  // stage 1: y = w0 (x) feature, 10x [512x1024x256] WMMA GEMM (20480 waves)
  gemm_y_kernel<<<20480 / 8, 256, 0, stream>>>(w0t, ft, y);
  // stage 2: sparse bilinear sample + bias + leaky
  sample_x_kernel<<<NPIX, 256, 0, stream>>>(y, iL, iR, swl, swr, b0, x);
  // stage 3: 1x1 conv [128x40960x512] WMMA GEMM (20480 waves)
  gemm_z_kernel<<<20480 / 8, 256, 0, stream>>>(w1b, x, b1, zpad);
  // stage 4: 3x3 conv as implicit WMMA GEMM, K=9*128 (20480 waves)
  conv3x3_kernel<<<20480 / 8, 256, 0, stream>>>(w2t, zpad, b2, cout);
  // stage 5: head + sigmoid
  head_kernel<<<(NPIX + 255) / 256, 256, 0, stream>>>(cout, w3, b3, out);
}
